// VarLSTM_52355651338788
// MI455X (gfx1250) — matrix operand order repaired and checked by reading, not compiled, not run on hardware
//
#include <hip/hip_runtime.h>

// ---------------------------------------------------------------------------
// VarLSTM for MI455X (gfx1250): per-timestep fused WMMA GEMM + gate kernel.
// Gate dimension permuted to q = 4*j + g so each 64-col tile holds complete
// i/f/g/o quadruples -> GEMM and pointwise LSTM update fuse in one kernel.
// f16 WMMA (V_WMMA_F32_16X16X32_F16), f32 accumulate.
// ---------------------------------------------------------------------------

#define B_   1024
#define T_   100
#define H_   256
#define G4   1024   // 4*H
#define D2   512    // 2*H

typedef _Float16 h8   __attribute__((ext_vector_type(8)));
typedef _Float16 v16h __attribute__((ext_vector_type(16)));
typedef float    v8f  __attribute__((ext_vector_type(8)));

// scratch layout (bytes)
static constexpr size_t OFF_M    = 0;                       // m[B] f32
static constexpr size_t OFF_XC   = 4096;                    // xcent[B][T] f32
static constexpr size_t OFF_BIAS = OFF_XC   + 409600;       // biasp[4][1024] f32
static constexpr size_t OFF_WIH1 = OFF_BIAS + 16384;        // wih1p[2][1024][2] f32
static constexpr size_t OFF_WHH  = OFF_WIH1 + 16384;        // whhp[4][1024][256] f16
static constexpr size_t OFF_WIH2 = OFF_WHH  + 2097152;      // wih2p[2][1024][512] f16
static constexpr size_t OFF_HM   = OFF_WIH2 + 2097152;      // hm[4][2][B][H] f16 (dbl-buf)
static constexpr size_t OFF_C    = OFF_HM   + 4194304;      // c[4][B][H] f32
static constexpr size_t OFF_PART = OFF_C    + 4194304;      // partial[T][B][32] f32
static constexpr size_t OFF_P1M  = OFF_PART + 13107200;     // p1m[T][B][512] f16

__device__ __forceinline__ float sigm_(float x) { return 1.0f / (1.0f + __expf(-x)); }

// ---------------------------------------------------------------------------
__global__ void prep_mean(const float* __restrict__ x0, float* m, float* xcent) {
    int b = blockIdx.x * blockDim.x + threadIdx.x;
    if (b >= B_) return;
    float s = 0.f;
    for (int t = 0; t < T_; ++t) s += x0[b * T_ + t];
    float mb = s * (1.0f / T_);
    m[b] = mb;
    for (int t = 0; t < T_; ++t) xcent[b * T_ + t] = x0[b * T_ + t] - mb;
}

// ---------------------------------------------------------------------------
__global__ void prep_weights(
    const float* Wih1f, const float* Whh1f, const float* bih1f, const float* bhh1f,
    const float* Wih1b, const float* Whh1b, const float* bih1b, const float* bhh1b,
    const float* Wih2f, const float* Whh2f, const float* bih2f, const float* bhh2f,
    const float* Wih2b, const float* Whh2b, const float* bih2b, const float* bhh2b,
    float* biasp, float* wih1p, _Float16* whhp, _Float16* wih2p)
{
    const float* Whh[4]  = {Whh1f, Whh1b, Whh2f, Whh2b};
    const float* Bih[4]  = {bih1f, bih1b, bih2f, bih2b};
    const float* Bhh[4]  = {bhh1f, bhh1b, bhh2f, bhh2b};
    const float* Wih1[2] = {Wih1f, Wih1b};
    const float* Wih2[2] = {Wih2f, Wih2b};
    int tid = blockIdx.x * blockDim.x + threadIdx.x;
    int nth = gridDim.x * blockDim.x;

    for (int idx = tid; idx < 4 * G4 * H_; idx += nth) {      // Whh gate-perm -> f16
        int l = idx / (G4 * H_), r = idx % (G4 * H_);
        int q = r / H_, k = r % H_;
        int j = q >> 2, g = q & 3;
        whhp[idx] = (_Float16)Whh[l][(g * H_ + j) * H_ + k];
    }
    for (int idx = tid; idx < 4 * G4; idx += nth) {           // bias perm
        int l = idx / G4, q = idx % G4;
        int j = q >> 2, g = q & 3;
        biasp[idx] = Bih[l][g * H_ + j] + Bhh[l][g * H_ + j];
    }
    for (int idx = tid; idx < 2 * G4 * 2; idx += nth) {       // Wih layer1 (2 cols, f32)
        int l = idx / (G4 * 2), r = idx % (G4 * 2);
        int q = r >> 1, cc = r & 1;
        int j = q >> 2, g = q & 3;
        wih1p[idx] = Wih1[l][(g * H_ + j) * 2 + cc];
    }
    for (int idx = tid; idx < 2 * G4 * D2; idx += nth) {      // Wih layer2 -> f16
        int l = idx / (G4 * D2), r = idx % (G4 * D2);
        int q = r / D2, d = r % D2;
        int j = q >> 2, g = q & 3;
        wih2p[idx] = (_Float16)Wih2[l][(g * H_ + j) * D2 + d];
    }
}

// ---------------------------------------------------------------------------
__global__ void init_state(_Float16* hm, float* c) {
    int tid = blockIdx.x * blockDim.x + threadIdx.x;
    int nth = gridDim.x * blockDim.x;
    for (int i = tid; i < 4 * 2 * B_ * H_; i += nth) hm[i] = (_Float16)0.0f;
    for (int i = tid; i < 4 * B_ * H_; i += nth)     c[i]  = 0.0f;
}

// ---------------------------------------------------------------------------
// One scan step of one layer, both directions (blockIdx.z). 64x64 tile of the
// gate preactivation matrix G[b, q] via WMMA, then fused gate nonlinearities.
// ---------------------------------------------------------------------------
template <int LAYER>
__global__ __launch_bounds__(256) void lstm_step(
    int s,
    const float* __restrict__ xcent, const float* __restrict__ m,
    const float* __restrict__ biasp, const float* __restrict__ wih1p,
    const _Float16* __restrict__ whhp, const _Float16* __restrict__ wih2p,
    _Float16* hm, float* c,
    const float* __restrict__ mask1, const float* __restrict__ mask2,
    const float* __restrict__ mask3, const float* __restrict__ mask4,
    const float* __restrict__ Wout,
    _Float16* p1m, float* partial)
{
    __shared__ __align__(16) _Float16 lA[64 * 40];
    __shared__ __align__(16) _Float16 lB[64 * 40];
    __shared__ float lG[64 * 68];
    __shared__ float lRed[256];

    const int dir   = blockIdx.z;
    const int tileX = blockIdx.x;          // q tile (64 cols -> 16 hidden units)
    const int tileY = blockIdx.y;          // b tile (64 rows)
    const int b0 = tileY * 64;
    const int q0 = tileX * 64;
    const int l  = (LAYER == 1) ? dir : 2 + dir;
    const int tin = (dir == 0) ? s : (T_ - 1 - s);

    const _Float16* hmR = hm + (size_t)(l * 2 + (s & 1)) * B_ * H_;
    _Float16*       hmW = hm + (size_t)(l * 2 + ((s + 1) & 1)) * B_ * H_;
    float*          cL  = c + (size_t)l * B_ * H_;
    const _Float16* whhL = whhp + (size_t)l * G4 * H_;

    const int tid  = threadIdx.x;
    const int lane = tid & 31;
    const int w    = tid >> 5;             // wave 0..7
    const int mrow = (w >> 1) * 16;        // wave's 16-row sub-tile
    const int ncol = (w & 1) * 32;         // wave's 32-col sub-tile

    v8f acc0 = {};
    v8f acc1 = {};

    const int KTOT = (LAYER == 1) ? H_ : (H_ + D2);
    const int grow = tid >> 2;             // staging row 0..63
    const int gseg = (tid & 3) * 8;        // 8-half segment within 32-chunk

    for (int kb = 0; kb < KTOT; kb += 32) {
        h8 av, bv;
        if (LAYER == 1 || kb < H_) {       // recurrent K: h*mask (f16), Whh perm
            av = *(const h8*)(hmR  + (size_t)(b0 + grow) * H_ + kb + gseg);
            bv = *(const h8*)(whhL + (size_t)(q0 + grow) * H_ + kb + gseg);
        } else {                           // layer-2 input K: p1*mask2, Wih2 perm
            int kk = kb - H_;
            av = *(const h8*)(p1m + ((size_t)tin * B_ + (b0 + grow)) * D2 + kk + gseg);
            bv = *(const h8*)(wih2p + ((size_t)dir * G4 + (q0 + grow)) * D2 + kk + gseg);
        }
        *(h8*)(lA + grow * 40 + gseg) = av;
        *(h8*)(lB + grow * 40 + gseg) = bv;
        __syncthreads();

        // A fragment: lane<16 -> M=lane, K 0..15; lane>=16 -> M=lane-16, K 16..31
        int arow = mrow + (lane & 15);
        int s0 = (lane & 16) ? 16 : 0;
        h8 alo = *(const h8*)(lA + arow * 40 + s0);
        h8 ahi = *(const h8*)(lA + arow * 40 + s0 + 8);
        v16h afrag = __builtin_shufflevector(alo, ahi, 0,1,2,3,4,5,6,7,8,9,10,11,12,13,14,15);

        // B fragments: lane<16 -> K 0..15, lane>=16 -> K 16..31; N = lane&15
        int kk0 = (lane & 16) ? 16 : 0;
        int brow0 = ncol + (lane & 15);
        h8 blo0 = *(const h8*)(lB + brow0 * 40 + kk0);
        h8 bhi0 = *(const h8*)(lB + brow0 * 40 + kk0 + 8);
        v16h bfrag0 = __builtin_shufflevector(blo0, bhi0, 0,1,2,3,4,5,6,7,8,9,10,11,12,13,14,15);
        int brow1 = brow0 + 16;
        h8 blo1 = *(const h8*)(lB + brow1 * 40 + kk0);
        h8 bhi1 = *(const h8*)(lB + brow1 * 40 + kk0 + 8);
        v16h bfrag1 = __builtin_shufflevector(blo1, bhi1, 0,1,2,3,4,5,6,7,8,9,10,11,12,13,14,15);

        acc0 = __builtin_amdgcn_wmma_f32_16x16x32_f16(false, afrag, false, bfrag0,
                                                      (short)0, acc0, false, false);
        acc1 = __builtin_amdgcn_wmma_f32_16x16x32_f16(false, afrag, false, bfrag1,
                                                      (short)0, acc1, false, false);
        __syncthreads();
    }

    // dump accumulators: lane<16 -> M=v, N=lane; lane>=16 -> M=8+v, N=lane-16
    {
        int rbase = mrow + ((lane & 16) ? 8 : 0);
        int cbase = ncol + (lane & 15);
#pragma unroll
        for (int v = 0; v < 8; ++v) {
            lG[(rbase + v) * 68 + cbase]      = acc0[v];
            lG[(rbase + v) * 68 + cbase + 16] = acc1[v];
        }
    }
    __syncthreads();

    // ---- fused pointwise LSTM update: thread -> (b_loc, 4 hidden units) ----
    const int b_loc = tid >> 2;
    const int grp   = tid & 3;
    const int b     = b0 + b_loc;
    const float* biasL = biasp + l * G4;
    float pd = 0.f;

#pragma unroll
    for (int u = 0; u < 4; ++u) {
        int j_loc = grp * 4 + u;               // 0..15 within tile
        int j = (q0 >> 2) + j_loc;             // hidden unit 0..255
        int qg = 4 * j_loc;
        float gi = lG[b_loc * 68 + qg + 0];
        float gf = lG[b_loc * 68 + qg + 1];
        float gg = lG[b_loc * 68 + qg + 2];
        float go = lG[b_loc * 68 + qg + 3];
        int qglob = q0 + qg;
        gi += biasL[qglob + 0]; gf += biasL[qglob + 1];
        gg += biasL[qglob + 2]; go += biasL[qglob + 3];
        if (LAYER == 1) {                      // rank-2 input contribution
            float xc = xcent[b * T_ + tin];
            float mb = m[b];
            const float* wp = wih1p + (size_t)dir * G4 * 2;
            gi += xc * wp[(qglob + 0) * 2] + mb * wp[(qglob + 0) * 2 + 1];
            gf += xc * wp[(qglob + 1) * 2] + mb * wp[(qglob + 1) * 2 + 1];
            gg += xc * wp[(qglob + 2) * 2] + mb * wp[(qglob + 2) * 2 + 1];
            go += xc * wp[(qglob + 3) * 2] + mb * wp[(qglob + 3) * 2 + 1];
        }
        float cold = cL[b * H_ + j];
        float cnew = sigm_(gf) * cold + sigm_(gi) * tanhf(gg);
        float hnew = sigm_(go) * tanhf(cnew);
        cL[b * H_ + j] = cnew;
        const float* mh = (LAYER == 1) ? mask1 : mask3;
        hmW[b * H_ + j] = (_Float16)(hnew * mh[b * H_ + j]);
        int d = j + 256 * dir;
        if (LAYER == 1) {
            p1m[((size_t)s * B_ + b) * D2 + d] = (_Float16)(hnew * mask2[b * D2 + d]);
        } else {
            pd += hnew * mask4[b * D2 + d] * Wout[d];
        }
    }
    if (LAYER == 2) {
        lRed[tid] = pd;
        __syncthreads();
        if (grp == 0) {
            float sum = lRed[tid] + lRed[tid + 1] + lRed[tid + 2] + lRed[tid + 3];
            partial[((size_t)s * B_ + b) * 32 + dir * 16 + tileX] = sum;
        }
    }
}

// ---------------------------------------------------------------------------
__global__ void finalize(const float* __restrict__ partial, const float* __restrict__ m,
                         const float* __restrict__ bout, float* out) {
    int idx = blockIdx.x * blockDim.x + threadIdx.x;
    if (idx >= B_ * T_) return;
    int b = idx / T_, t = idx % T_;
    const float* p = partial + ((size_t)t * B_ + b) * 32;
    float s = 0.f;
#pragma unroll
    for (int r = 0; r < 32; ++r) s += p[r];
    out[idx] = s + bout[0] + m[b];
}

// ---------------------------------------------------------------------------
extern "C" void kernel_launch(void* const* d_in, const int* in_sizes, int n_in,
                              void* d_out, int out_size, void* d_ws, size_t ws_size,
                              hipStream_t stream) {
    const float* x0    = (const float*)d_in[0];
    const float* Wih1f = (const float*)d_in[1];
    const float* Whh1f = (const float*)d_in[2];
    const float* bih1f = (const float*)d_in[3];
    const float* bhh1f = (const float*)d_in[4];
    const float* Wih1b = (const float*)d_in[5];
    const float* Whh1b = (const float*)d_in[6];
    const float* bih1b = (const float*)d_in[7];
    const float* bhh1b = (const float*)d_in[8];
    const float* Wih2f = (const float*)d_in[9];
    const float* Whh2f = (const float*)d_in[10];
    const float* bih2f = (const float*)d_in[11];
    const float* bhh2f = (const float*)d_in[12];
    const float* Wih2b = (const float*)d_in[13];
    const float* Whh2b = (const float*)d_in[14];
    const float* bih2b = (const float*)d_in[15];
    const float* bhh2b = (const float*)d_in[16];
    const float* Wout  = (const float*)d_in[17];
    const float* bout  = (const float*)d_in[18];
    const float* mask1 = (const float*)d_in[19];
    const float* mask2 = (const float*)d_in[20];
    const float* mask3 = (const float*)d_in[21];
    const float* mask4 = (const float*)d_in[22];

    char* ws = (char*)d_ws;
    float*     m_     = (float*)(ws + OFF_M);
    float*     xcent  = (float*)(ws + OFF_XC);
    float*     biasp  = (float*)(ws + OFF_BIAS);
    float*     wih1p  = (float*)(ws + OFF_WIH1);
    _Float16*  whhp   = (_Float16*)(ws + OFF_WHH);
    _Float16*  wih2p  = (_Float16*)(ws + OFF_WIH2);
    _Float16*  hm     = (_Float16*)(ws + OFF_HM);
    float*     c_     = (float*)(ws + OFF_C);
    float*     part   = (float*)(ws + OFF_PART);
    _Float16*  p1m    = (_Float16*)(ws + OFF_P1M);

    prep_mean<<<4, 256, 0, stream>>>(x0, m_, xcent);
    prep_weights<<<256, 256, 0, stream>>>(Wih1f, Whh1f, bih1f, bhh1f,
                                          Wih1b, Whh1b, bih1b, bhh1b,
                                          Wih2f, Whh2f, bih2f, bhh2f,
                                          Wih2b, Whh2b, bih2b, bhh2b,
                                          biasp, wih1p, whhp, wih2p);
    init_state<<<1024, 256, 0, stream>>>(hm, c_);

    dim3 grid(16, 16, 2);
    for (int s = 0; s < T_; ++s)
        lstm_step<1><<<grid, 256, 0, stream>>>(s, xcent, m_, biasp, wih1p, whhp, wih2p,
                                               hm, c_, mask1, mask2, mask3, mask4, Wout,
                                               p1m, part);
    for (int s = 0; s < T_; ++s)
        lstm_step<2><<<grid, 256, 0, stream>>>(s, xcent, m_, biasp, wih1p, whhp, wih2p,
                                               hm, c_, mask1, mask2, mask3, mask4, Wout,
                                               p1m, part);

    finalize<<<400, 256, 0, stream>>>(part, m_, bout, (float*)d_out);
}
